// GCNN_net_29463475650824
// MI455X (gfx1250) — matrix operand order repaired and checked by reading, not compiled, not run on hardware
//
#include <hip/hip_runtime.h>
#include <hip/hip_bf16.h>

// ---------------------------------------------------------------------------
// TransformerConv x2 + BN/ReLU for gfx1250 (MI455X).
// Dense GEMMs use v_wmma_f32_16x16x32_bf16 (wave32, 16x16 tiles, f32 accum)
// with 1x4 N-register-blocking: one A fragment feeds 4 WMMAs.
// Edge attention uses wave-per-edge float4 gathers + f32 atomics for the
// segment softmax / scatter-add.
// ---------------------------------------------------------------------------

#define GN  10000      // nodes
#define GE  160000     // edges
#define GH  2          // heads
#define GC  128        // per-head channels
#define GD  256        // H*C
#define GEPS 1e-5f

typedef __bf16 bf16_t;
typedef __attribute__((ext_vector_type(16))) __bf16 v16bf;
typedef __attribute__((ext_vector_type(8)))  __bf16 v8bf;
typedef __attribute__((ext_vector_type(8)))  float  v8f;

// ---------------------------------------------------------------- utilities
__global__ void fill_f32(float* __restrict__ p, float v, int n) {
    int i = blockIdx.x * blockDim.x + threadIdx.x;
    if (i < n) p[i] = v;
}

__global__ void cast_f32_to_bf16(const float* __restrict__ src,
                                 bf16_t* __restrict__ dst, int n) {
    int i = blockIdx.x * blockDim.x + threadIdx.x;
    if (i < n) dst[i] = (bf16_t)src[i];
}

// WT[n*K + k] = (bf16) W[k*Nc + n]   (row-major [K,Nc] -> transposed bf16 [Nc,K])
__global__ void wcast_transpose(const float* __restrict__ W,
                                bf16_t* __restrict__ WT, int K, int Nc) {
    int i = blockIdx.x * blockDim.x + threadIdx.x;
    if (i >= K * Nc) return;
    int n = i / K;
    int k = i - n * K;
    WT[i] = (bf16_t)W[k * Nc + n];
}

__device__ __forceinline__ void atomicMaxF32(float* addr, float v) {
    if (v >= 0.0f) atomicMax((int*)addr, __float_as_int(v));
    else           atomicMin((unsigned int*)addr, __float_as_uint(v));
}

// ------------------------------------------------------------ bf16 WMMA GEMM
// C[M,Nc] = A[M,K] @ B[K,Nc] + bias,  B supplied transposed (BT[Nc,K]).
// One wave per 16x64 output strip (4 accumulators); A fragment reused 4x.
// K multiple of 32; M multiple of 16; Nc multiple of 64.
__global__ __launch_bounds__(32)
void gemm_bf16_wmma(const bf16_t* __restrict__ A,
                    const bf16_t* __restrict__ BT,
                    const float* __restrict__ bias,
                    float* __restrict__ C,
                    int M, int K, int Nc) {
    const int lane = threadIdx.x & 31;
    const int half = lane >> 4;     // 0: lanes 0-15, 1: lanes 16-31
    const int ml   = lane & 15;
    const int m0   = blockIdx.x * 16;
    const int n0   = blockIdx.y * 64;

    const bf16_t* Arow = A + (size_t)(m0 + ml) * K;   // A-frag row = lane&15
    const bf16_t* Brow0 = BT + (size_t)(n0 + ml) * K; // B-frag col = lane&15
    const size_t bstride = (size_t)16 * K;            // next 16-col group

    v8f acc0 = {}, acc1 = {}, acc2 = {}, acc3 = {};
    for (int kk = 0; kk < K; kk += 32) {
        // A 16x32 bf16 layout (contiguous, matches B): lane(half=0) K=[0..15],
        // lane(half=1) K=[16..31]; two contiguous 16B chunks.
        const bf16_t* a0 = Arow + kk + half * 16;
        v8bf alo = *(const v8bf*)(a0);
        v8bf ahi = *(const v8bf*)(a0 + 8);
        v16bf afrag = __builtin_shufflevector(alo, ahi,
            0,1,2,3,4,5,6,7,8,9,10,11,12,13,14,15);

        // B 32x16 bf16 from BT: lane holds 16 contiguous K (32B load).
        const bf16_t* b0 = Brow0 + kk + half * 16;
        v16bf bf0 = *(const v16bf*)(b0);
        v16bf bf1 = *(const v16bf*)(b0 + bstride);
        v16bf bf2 = *(const v16bf*)(b0 + 2 * bstride);
        v16bf bf3 = *(const v16bf*)(b0 + 3 * bstride);

        acc0 = __builtin_amdgcn_wmma_f32_16x16x32_bf16(
            false, afrag, false, bf0, (short)0, acc0, false, false);
        acc1 = __builtin_amdgcn_wmma_f32_16x16x32_bf16(
            false, afrag, false, bf1, (short)0, acc1, false, false);
        acc2 = __builtin_amdgcn_wmma_f32_16x16x32_bf16(
            false, afrag, false, bf2, (short)0, acc2, false, false);
        acc3 = __builtin_amdgcn_wmma_f32_16x16x32_bf16(
            false, afrag, false, bf3, (short)0, acc3, false, false);
    }

    // C/D layout: col n = lane&15, row m = half*8 + r for VGPR r.
    v8f accs[4] = { acc0, acc1, acc2, acc3 };
#pragma unroll
    for (int j = 0; j < 4; ++j) {
        const int nc = n0 + j * 16 + ml;
        const float bc = bias ? bias[nc] : 0.0f;
        float* Cc = C + (size_t)(m0 + half * 8) * Nc + nc;
#pragma unroll
        for (int r = 0; r < 8; ++r)
            Cc[(size_t)r * Nc] = accs[j][r] + bc;
    }
}

// ----------------------------------------------------------- edge kernels
// Pass 1: alpha_raw[e][h] = (Q[dst] . (K[src] + ea*We)) / sqrt(C); atomic max.
__global__ __launch_bounds__(256)
void edge_alpha(const float* __restrict__ Q, const float* __restrict__ Kf,
                const float* __restrict__ We, const float* __restrict__ ea,
                const int* __restrict__ src, const int* __restrict__ dst,
                float* __restrict__ alpha, float* __restrict__ mbuf) {
    const int wave = (blockIdx.x * blockDim.x + threadIdx.x) >> 5;
    const int lane = threadIdx.x & 31;
    if (wave >= GE) return;
    const int s = src[wave], d = dst[wave];
    const float a = ea[wave];
    const float scale = 0.08838834764831845f;  // 1/sqrt(128)

#pragma unroll
    for (int h = 0; h < GH; ++h) {
        const int base = h * GC + lane * 4;
        float4 qv = *(const float4*)(Q  + (size_t)d * GD + base);
        float4 kv = *(const float4*)(Kf + (size_t)s * GD + base);
        float4 wv = *(const float4*)(We + base);
        float acc = qv.x * (kv.x + a * wv.x) + qv.y * (kv.y + a * wv.y)
                  + qv.z * (kv.z + a * wv.z) + qv.w * (kv.w + a * wv.w);
#pragma unroll
        for (int off = 16; off > 0; off >>= 1)
            acc += __shfl_xor(acc, off, 32);
        if (lane == 0) {
            float v = acc * scale;
            alpha[(size_t)wave * GH + h] = v;
            atomicMaxF32(&mbuf[(size_t)d * GH + h], v);
        }
    }
}

// Pass 2: ex = exp(alpha - m[dst]); atomic segment sum.
__global__ void edge_exp(float* __restrict__ alpha,
                         const float* __restrict__ mbuf,
                         float* __restrict__ sbuf,
                         const int* __restrict__ dst) {
    int i = blockIdx.x * blockDim.x + threadIdx.x;
    if (i >= GE * GH) return;
    int e = i >> 1, h = i & 1;
    int d = dst[e];
    float ex = __expf(alpha[i] - mbuf[d * GH + h]);
    alpha[i] = ex;
    atomicAdd(&sbuf[d * GH + h], ex);
}

// Pass 3: out[dst] += (V[src] + ea*We) * (ex / (s[dst]+1e-16)).
__global__ __launch_bounds__(256)
void edge_scatter(const float* __restrict__ Vf, const float* __restrict__ We,
                  const float* __restrict__ ea, const float* __restrict__ exb,
                  const float* __restrict__ sbuf,
                  const int* __restrict__ src, const int* __restrict__ dst,
                  float* __restrict__ out) {
    const int wave = (blockIdx.x * blockDim.x + threadIdx.x) >> 5;
    const int lane = threadIdx.x & 31;
    if (wave >= GE) return;
    const int s = src[wave], d = dst[wave];
    const float a = ea[wave];

#pragma unroll
    for (int h = 0; h < GH; ++h) {
        const float w = exb[(size_t)wave * GH + h]
                      / (sbuf[(size_t)d * GH + h] + 1e-16f);
        const int base = h * GC + lane * 4;
        float4 vv = *(const float4*)(Vf + (size_t)s * GD + base);
        float4 wv = *(const float4*)(We + base);
        float* o = out + (size_t)d * GD + base;
        atomicAdd(o + 0, (vv.x + a * wv.x) * w);
        atomicAdd(o + 1, (vv.y + a * wv.y) * w);
        atomicAdd(o + 2, (vv.z + a * wv.z) * w);
        atomicAdd(o + 3, (vv.w + a * wv.w) * w);
    }
}

// --------------------------------------------------------------- batch norm
#define BN_ROWS 64
__global__ __launch_bounds__(GD)
void bn_reduce(const float* __restrict__ X, float* __restrict__ sum,
               float* __restrict__ sq) {
    const int c  = threadIdx.x;            // channel (0..255), coalesced reads
    const int r0 = blockIdx.x * BN_ROWS;
    float s = 0.f, q = 0.f;
    for (int r = 0; r < BN_ROWS; ++r) {
        int rr = r0 + r;
        if (rr < GN) {
            float x = X[(size_t)rr * GD + c];
            s += x; q += x * x;
        }
    }
    atomicAdd(&sum[c], s);
    atomicAdd(&sq[c],  q);
}

__global__ void bn_relu_apply(const float* __restrict__ X,
                              const float* __restrict__ sum,
                              const float* __restrict__ sq,
                              const float* __restrict__ g,
                              const float* __restrict__ b,
                              float* __restrict__ outF,
                              bf16_t* __restrict__ outB) {
    int i = blockIdx.x * blockDim.x + threadIdx.x;
    if (i >= GN * GD) return;
    int c = i & (GD - 1);
    const float invN = 1.0f / (float)GN;
    float mu  = sum[c] * invN;
    float var = sq[c] * invN - mu * mu;
    float y = (X[i] - mu) * rsqrtf(var + GEPS) * g[c] + b[c];
    y = fmaxf(y, 0.0f);
    if (outF) outF[i] = y;
    if (outB) outB[i] = (bf16_t)y;
}

// ---------------------------------------------------------------- host side
extern "C" void kernel_launch(void* const* d_in, const int* in_sizes, int n_in,
                              void* d_out, int out_size, void* d_ws, size_t ws_size,
                              hipStream_t stream) {
    (void)in_sizes; (void)n_in; (void)out_size; (void)ws_size;

    const float* x     = (const float*)d_in[0];
    const int*   eidx  = (const int*)d_in[1];
    const float* eattr = (const float*)d_in[2];
    const int* src = eidx;
    const int* dst = eidx + GE;

    // per-layer parameter tables (dict order)
    const float* Wq[2] = { (const float*)d_in[3],  (const float*)d_in[14] };
    const float* bq[2] = { (const float*)d_in[4],  (const float*)d_in[15] };
    const float* Wk[2] = { (const float*)d_in[5],  (const float*)d_in[16] };
    const float* bk[2] = { (const float*)d_in[6],  (const float*)d_in[17] };
    const float* Wv[2] = { (const float*)d_in[7],  (const float*)d_in[18] };
    const float* bv[2] = { (const float*)d_in[8],  (const float*)d_in[19] };
    const float* We[2] = { (const float*)d_in[9],  (const float*)d_in[20] };
    const float* Ws[2] = { (const float*)d_in[10], (const float*)d_in[21] };
    const float* bs[2] = { (const float*)d_in[11], (const float*)d_in[22] };
    const float* gg[2] = { (const float*)d_in[12], (const float*)d_in[23] };
    const float* bb[2] = { (const float*)d_in[13], (const float*)d_in[24] };

    // workspace carve-out (256B aligned)
    size_t off = 0;
    auto carve = [&](size_t bytes) -> void* {
        void* p = (char*)d_ws + off;
        off += (bytes + 255) & ~(size_t)255;
        return p;
    };
    bf16_t* xb    = (bf16_t*)carve((size_t)GN * GD * sizeof(bf16_t));
    bf16_t* wt    = (bf16_t*)carve((size_t)GD * GD * sizeof(bf16_t));
    float*  qb    = (float*)carve((size_t)GN * GD * sizeof(float));
    float*  kb    = (float*)carve((size_t)GN * GD * sizeof(float));
    float*  vb    = (float*)carve((size_t)GN * GD * sizeof(float));
    float*  accb  = (float*)carve((size_t)GN * GD * sizeof(float));
    float*  alphb = (float*)carve((size_t)GE * GH * sizeof(float));
    float*  mb    = (float*)carve((size_t)GN * GH * sizeof(float));
    float*  sb    = (float*)carve((size_t)GN * GH * sizeof(float));
    float*  bnsum = (float*)carve(GD * sizeof(float));
    float*  bnsq  = (float*)carve(GD * sizeof(float));

    const int T = 256;
    auto blocks = [](long n, int t) { return (int)((n + t - 1) / t); };

    // layer 1 GEMM input: cast x (N x 128) to bf16
    cast_f32_to_bf16<<<blocks((long)GN * 128, T), T, 0, stream>>>(x, xb, GN * 128);

    for (int L = 0; L < 2; ++L) {
        const int K = (L == 0) ? 128 : GD;          // 128 then 256

        const float* Wmat[4] = { Wq[L], Wk[L], Wv[L], Ws[L] };
        const float* Bvec[4] = { bq[L], bk[L], bv[L], bs[L] };
        float*       Dest[4] = { qb, kb, vb, accb };

        for (int i = 0; i < 4; ++i) {
            wcast_transpose<<<blocks((long)K * GD, T), T, 0, stream>>>(
                Wmat[i], wt, K, GD);
            dim3 grid(GN / 16, GD / 64);   // 1x4 N-blocked waves
            gemm_bf16_wmma<<<grid, 32, 0, stream>>>(
                xb, wt, Bvec[i], Dest[i], GN, K, GD);
        }

        // segment softmax state
        fill_f32<<<blocks((long)GN * GH, T), T, 0, stream>>>(mb, -INFINITY, GN * GH);
        fill_f32<<<blocks((long)GN * GH, T), T, 0, stream>>>(sb, 0.0f, GN * GH);

        // edge passes (8 waves / block, wave per edge)
        const int eb = blocks((long)GE * 32, T);
        edge_alpha<<<eb, T, 0, stream>>>(qb, kb, We[L], eattr, src, dst, alphb, mb);
        edge_exp<<<blocks((long)GE * GH, T), T, 0, stream>>>(alphb, mb, sb, dst);
        edge_scatter<<<eb, T, 0, stream>>>(vb, We[L], eattr, alphb, sb, src, dst, accb);

        // BN + ReLU (fused bf16 recast for next layer's GEMM input)
        fill_f32<<<1, GD, 0, stream>>>(bnsum, 0.0f, GD);
        fill_f32<<<1, GD, 0, stream>>>(bnsq,  0.0f, GD);
        bn_reduce<<<blocks(GN, BN_ROWS), GD, 0, stream>>>(accb, bnsum, bnsq);

        float*  outF = (L == 1) ? (float*)d_out : nullptr;
        bf16_t* outB = (L == 0) ? xb : nullptr;
        bn_relu_apply<<<blocks((long)GN * GD, T), T, 0, stream>>>(
            accb, bnsum, bnsq, gg[L], bb[L], outF, outB);
    }
}
